// MyMlp_39281770889905
// MI455X (gfx1250) — matrix operand order repaired and checked by reading, not compiled, not run on hardware
//
#include <hip/hip_runtime.h>
#include <hip/hip_bf16.h>
#include <math.h>

// ---------------- CDNA5 WMMA fragment types ----------------
typedef __attribute__((ext_vector_type(16))) _Float16 f16x16;
typedef __attribute__((ext_vector_type(8)))  _Float16 f16x8;
typedef __attribute__((ext_vector_type(8)))  float    f32x8;

// ---------------- problem dims (match reference setup_inputs) ----------------
static constexpr int B_   = 8;
static constexpr int N_   = 3136;
static constexpr int CIN  = 256;
static constexpr int CHID = 1024;
static constexpr int H_   = 56;
static constexpr int W_   = 56;
static constexpr int HW   = H_ * W_;
static constexpr int BN   = B_ * N_;
static constexpr int CP1  = CHID + 1;

__device__ __forceinline__ int iclamp(int v, int lo, int hi) {
  return v < lo ? lo : (v > hi ? hi : v);
}

// ---------------- f32 -> f16 convert (grid-stride) ----------------
__global__ void cvt_f32_f16(const float* __restrict__ in, _Float16* __restrict__ out, int n) {
  int i = blockIdx.x * blockDim.x + threadIdx.x;
  int stride = gridDim.x * blockDim.x;
  for (; i < n; i += stride) out[i] = (_Float16)in[i];
}

// ---------------- zero fill ----------------
__global__ void zero_f32(float* __restrict__ p, int n) {
  int i = blockIdx.x * blockDim.x + threadIdx.x;
  int stride = gridDim.x * blockDim.x;
  for (; i < n; i += stride) p[i] = 0.0f;
}

// ---------------- WMMA fragment loads (CDNA5 ISA 7.12.2 layouts) ----------------
// A: 16-bit 16x32, lane m = lane&15, hi = lane>>4.
//    halfs 0..7  hold K = hi*8   + t   (VGPR0..3)
//    halfs 8..15 hold K = 16+hi*8 + t  (VGPR4..7)
__device__ __forceinline__ f16x16 load_fragA(const _Float16* __restrict__ rowp, int k, int hi) {
  f16x8 r0 = *(const f16x8*)(rowp + k + hi * 8);
  f16x8 r1 = *(const f16x8*)(rowp + k + 16 + hi * 8);
  f16x16 f;
#pragma unroll
  for (int t = 0; t < 8; ++t) { f[t] = r0[t]; f[8 + t] = r1[t]; }
  return f;
}
// B: 32x16, lane n = lane&15, hi = lane>>4; halfs 0..7 hold K = hi*8 + t, halfs 8..15 hold K = 16+hi*8 + t.
__device__ __forceinline__ f16x16 load_fragB(const _Float16* __restrict__ rowp, int k, int hi) {
  f16x8 r0 = *(const f16x8*)(rowp + k + hi * 8);
  f16x8 r1 = *(const f16x8*)(rowp + k + 16 + hi * 8);
  f16x16 f;
#pragma unroll
  for (int t = 0; t < 8; ++t) { f[t] = r0[t]; f[8 + t] = r1[t]; }
  return f;
}

// ---------------- WMMA GEMM: C[M,N] = act(A[M,K] @ B[N,K]^T + bias[N]) ----------------
// block = 256 threads = 8 waves (4 along M x 2 along N); wave tile 32x32 (2x2 wmma);
// block tile 128(M) x 64(N). M%128==0, N%64==0, K%32==0 guaranteed by caller.
template <bool RELU>
__global__ __launch_bounds__(256) void wmma_gemm_f16(
    const _Float16* __restrict__ A, const _Float16* __restrict__ Bw,
    const float* __restrict__ bias, float* __restrict__ C,
    int M, int N, int K) {
  const int lane = threadIdx.x & 31;
  const int wid  = threadIdx.x >> 5;
  const int wm   = wid & 3;
  const int wn   = wid >> 2;
  const int m0   = blockIdx.y * 128 + wm * 32;
  const int n0   = blockIdx.x * 64 + wn * 32;
  const int row  = lane & 15;
  const int hi   = lane >> 4;

  const _Float16* a0 = A + (size_t)(m0 + row) * K;
  const _Float16* a1 = A + (size_t)(m0 + 16 + row) * K;
  const _Float16* b0 = Bw + (size_t)(n0 + row) * K;
  const _Float16* b1 = Bw + (size_t)(n0 + 16 + row) * K;

  f32x8 acc00 = {}, acc01 = {}, acc10 = {}, acc11 = {};

  for (int k = 0; k < K; k += 32) {
    if (k + 32 < K) {  // uniform branch: prefetch next K tile (global_prefetch_b8)
      __builtin_prefetch(a0 + k + 32, 0, 0);
      __builtin_prefetch(a1 + k + 32, 0, 0);
      __builtin_prefetch(b0 + k + 32, 0, 0);
      __builtin_prefetch(b1 + k + 32, 0, 0);
    }
    f16x16 fa0 = load_fragA(a0, k, hi);
    f16x16 fa1 = load_fragA(a1, k, hi);
    f16x16 fb0 = load_fragB(b0, k, hi);
    f16x16 fb1 = load_fragB(b1, k, hi);
    acc00 = __builtin_amdgcn_wmma_f32_16x16x32_f16(false, fa0, false, fb0, (short)0, acc00, false, false);
    acc01 = __builtin_amdgcn_wmma_f32_16x16x32_f16(false, fa0, false, fb1, (short)0, acc01, false, false);
    acc10 = __builtin_amdgcn_wmma_f32_16x16x32_f16(false, fa1, false, fb0, (short)0, acc10, false, false);
    acc11 = __builtin_amdgcn_wmma_f32_16x16x32_f16(false, fa1, false, fb1, (short)0, acc11, false, false);
  }

  // D layout: VGPR r -> m = r + 8*hi ; n = lane&15
  const int nc0 = n0 + (lane & 15);
  const float bv0 = bias[nc0];
  const float bv1 = bias[nc0 + 16];
#pragma unroll
  for (int r = 0; r < 8; ++r) {
    const int mA = m0 + r + 8 * hi;
    const int mB = mA + 16;
    float v;
    v = acc00[r] + bv0; if (RELU) v = fmaxf(v, 0.f); C[(size_t)mA * N + nc0] = v;
    v = acc01[r] + bv1; if (RELU) v = fmaxf(v, 0.f); C[(size_t)mA * N + nc0 + 16] = v;
    v = acc10[r] + bv0; if (RELU) v = fmaxf(v, 0.f); C[(size_t)mB * N + nc0] = v;
    v = acc11[r] + bv1; if (RELU) v = fmaxf(v, 0.f); C[(size_t)mB * N + nc0 + 16] = v;
  }
}

// ---------------- token2map scatter: one block per token ----------------
__global__ void token2map_scatter(const float* __restrict__ h, const float* __restrict__ loc,
                                  float* __restrict__ map) {
  const int t = blockIdx.x;              // 0..BN-1
  const int b = t / N_;
  float lx = fminf(fmaxf(loc[2 * t], -1.f), 1.f);
  float ly = fminf(fmaxf(loc[2 * t + 1], -1.f), 1.f);
  int px = iclamp((int)rintf(0.5f * (lx + 1.f) * (float)W_ - 0.5f), 0, W_ - 1);
  int py = iclamp((int)rintf(0.5f * (ly + 1.f) * (float)H_ - 0.5f), 0, H_ - 1);
  const size_t base = ((size_t)b * HW + (size_t)py * W_ + px) * CP1;
  const float* src = h + (size_t)t * CHID;
  for (int c = threadIdx.x; c < CHID; c += blockDim.x)
    atomicAdd(&map[base + c], src[c]);
  if (threadIdx.x == 0) atomicAdd(&map[base + CHID], 1.0f);
}

// ---------------- normalize: feat /= (mask+eps), mask -> binary ----------------
__global__ void map_normalize(float* __restrict__ map) {
  float* rowp = map + (size_t)blockIdx.x * CP1;
  const float m = rowp[CHID];
  const float bin = (m > 0.f) ? 1.f : 0.f;
  const float inv = bin / (m + 1e-6f);
  __syncthreads();  // everyone read m before lane0 rewrites it
  for (int c = threadIdx.x; c < CHID; c += blockDim.x) rowp[c] *= inv;
  if (threadIdx.x == 0) rowp[CHID] = bin;
}

// ---------------- gaussian hole fill (k=3, sigma=2) ----------------
__global__ void smooth_fill(const float* __restrict__ map, float* __restrict__ feat2) {
  const int p = blockIdx.x;             // b*HW + pix
  const int b = p / HW;
  const int pix = p - b * HW;
  const int y = pix / W_, x = pix - y * W_;
  __shared__ float s_w[9];
  __shared__ int s_nb[9];
  __shared__ float s_smm, s_cm;
  if (threadIdx.x == 0) {
    const float g = 0.8824969025845955f;  // exp(-1/(2*sigma^2)), sigma=2
    const float s1 = 1.f + 2.f * g;
    const float norm = 1.f / (s1 * s1);
    float smm = 0.f;
    for (int dy = -1; dy <= 1; ++dy)
      for (int dx = -1; dx <= 1; ++dx) {
        const int k = (dy + 1) * 3 + (dx + 1);
        const int yy = y + dy, xx = x + dx;
        const bool ok = (yy >= 0) & (yy < H_) & (xx >= 0) & (xx < W_);
        const float wv = ((dy == 0) ? 1.f : g) * ((dx == 0) ? 1.f : g) * norm;
        const int nb = ok ? (b * HW + yy * W_ + xx) : p;
        s_w[k] = ok ? wv : 0.f;
        s_nb[k] = nb;
        if (ok) smm += wv * map[(size_t)nb * CP1 + CHID];
      }
    s_smm = smm;
    s_cm = map[(size_t)p * CP1 + CHID];
  }
  __syncthreads();
  const float* crow = map + (size_t)p * CP1;
  float* orow = feat2 + (size_t)p * CHID;
  if (s_cm > 0.f) {  // occupied pixel: feat passes through
    for (int c = threadIdx.x; c < CHID; c += blockDim.x) orow[c] = crow[c];
  } else {           // hole: feat==0 here, out = fi * (sm_mask>0)
    const float smm = s_smm;
    const float inv = (smm > 0.f) ? 1.f / (smm + 1e-6f) : 0.f;
    for (int c = threadIdx.x; c < CHID; c += blockDim.x) {
      float s = 0.f;
#pragma unroll
      for (int k = 0; k < 9; ++k) s += s_w[k] * map[(size_t)s_nb[k] * CP1 + c];
      orow[c] = s * inv;
    }
  }
}

// ---------------- depthwise 3x3 conv + bias (channel-last) ----------------
__global__ void dwconv3x3(const float* __restrict__ fin, const float* __restrict__ wgt,
                          const float* __restrict__ bias, float* __restrict__ fout) {
  const int p = blockIdx.x;
  const int b = p / HW;
  const int pix = p - b * HW;
  const int y = pix / W_, x = pix - y * W_;
  __shared__ int s_nb[9];
  __shared__ float s_ok[9];
  if (threadIdx.x == 0) {
    for (int dy = -1; dy <= 1; ++dy)
      for (int dx = -1; dx <= 1; ++dx) {
        const int k = (dy + 1) * 3 + (dx + 1);
        const int yy = y + dy, xx = x + dx;
        const bool ok = (yy >= 0) & (yy < H_) & (xx >= 0) & (xx < W_);
        s_nb[k] = ok ? (b * HW + yy * W_ + xx) : p;
        s_ok[k] = ok ? 1.f : 0.f;
      }
  }
  __syncthreads();
  for (int c = threadIdx.x; c < CHID; c += blockDim.x) {
    float acc = bias[c];
#pragma unroll
    for (int k = 0; k < 9; ++k)
      acc += s_ok[k] * wgt[c * 9 + k] * fin[(size_t)s_nb[k] * CHID + c];
    fout[(size_t)p * CHID + c] = acc;
  }
}

// ---------------- map2token (bilinear) + exact GELU, emit f16 ----------------
__global__ void map2token_gelu(const float* __restrict__ fm, const float* __restrict__ loc,
                               _Float16* __restrict__ t16) {
  const int t = blockIdx.x;  // token
  const int b = t / N_;
  const float gx = (loc[2 * t] + 1.f) * (float)W_ * 0.5f - 0.5f;
  const float gy = (loc[2 * t + 1] + 1.f) * (float)H_ * 0.5f - 0.5f;
  const float x0f = floorf(gx), y0f = floorf(gy);
  const float wx1 = gx - x0f, wy1 = gy - y0f;
  const int x0 = (int)x0f, y0 = (int)y0f;

  float w[4];
  size_t idx[4];
  const int cx[4] = {x0, x0 + 1, x0, x0 + 1};
  const int cy[4] = {y0, y0, y0 + 1, y0 + 1};
  const float cw[4] = {(1.f - wx1) * (1.f - wy1), wx1 * (1.f - wy1),
                       (1.f - wx1) * wy1, wx1 * wy1};
#pragma unroll
  for (int k = 0; k < 4; ++k) {
    const bool valid = (cx[k] >= 0) & (cx[k] < W_) & (cy[k] >= 0) & (cy[k] < H_);
    const int lin = iclamp(cx[k], 0, W_ - 1) + iclamp(cy[k], 0, H_ - 1) * W_;
    idx[k] = ((size_t)b * HW + lin) * CHID;
    w[k] = valid ? cw[k] : 0.f;
  }
  _Float16* orow = t16 + (size_t)t * CHID;
  for (int c = threadIdx.x; c < CHID; c += blockDim.x) {
    float v = w[0] * fm[idx[0] + c] + w[1] * fm[idx[1] + c] +
              w[2] * fm[idx[2] + c] + w[3] * fm[idx[3] + c];
    v = 0.5f * v * (1.f + erff(v * 0.7071067811865475f));  // exact GELU
    orow[c] = (_Float16)v;
  }
}

// ---------------- host-side launch ----------------
extern "C" void kernel_launch(void* const* d_in, const int* in_sizes, int n_in,
                              void* d_out, int out_size, void* d_ws, size_t ws_size,
                              hipStream_t stream) {
  const float* x     = (const float*)d_in[0];  // [B,N,CIN]
  const float* loc   = (const float*)d_in[1];  // [B,N,2]
  const float* fc1_w = (const float*)d_in[2];  // [CHID,CIN]
  const float* fc1_b = (const float*)d_in[3];  // [CHID]
  const float* fc2_w = (const float*)d_in[4];  // [CIN,CHID]
  const float* fc2_b = (const float*)d_in[5];  // [CIN]
  const float* dw_w  = (const float*)d_in[6];  // [CHID,1,3,3]
  const float* dw_b  = (const float*)d_in[7];  // [CHID]
  // d_in[8..11] = H, W, kernel_size, sigma (device scalars; dims fixed by setup_inputs)
  float* out = (float*)d_out;

  // ---- carve workspace ----
  char* ws = (char*)d_ws;
  auto carve = [&](size_t bytes) {
    char* p = ws;
    ws += (bytes + 255) & ~(size_t)255;
    return p;
  };
  _Float16* x16  = (_Float16*)carve(sizeof(_Float16) * (size_t)BN * CIN);
  _Float16* w116 = (_Float16*)carve(sizeof(_Float16) * (size_t)CHID * CIN);
  _Float16* w216 = (_Float16*)carve(sizeof(_Float16) * (size_t)CIN * CHID);
  float* bh    = (float*)carve(sizeof(float) * (size_t)BN * CHID);   // fc1 out
  float* bmap  = (float*)carve(sizeof(float) * (size_t)B_ * HW * CP1);
  float* bfeat = (float*)carve(sizeof(float) * (size_t)B_ * HW * CHID);
  _Float16* t16 = (_Float16*)bh;  // alias: h is dead after scatter
  float* conv   = bmap;           // alias: map dead after smooth; conv stride = CHID

  const dim3 blk(256);

  // precision staging
  cvt_f32_f16<<<2048, blk, 0, stream>>>(x, x16, BN * CIN);
  cvt_f32_f16<<<256, blk, 0, stream>>>(fc1_w, w116, CHID * CIN);
  cvt_f32_f16<<<256, blk, 0, stream>>>(fc2_w, w216, CIN * CHID);
  zero_f32<<<4096, blk, 0, stream>>>(bmap, B_ * HW * CP1);

  // fc1: h = relu(x @ fc1_w^T + b)   [25088 x 1024], K=256
  wmma_gemm_f16<true><<<dim3(CHID / 64, BN / 128), blk, 0, stream>>>(
      x16, w116, fc1_b, bh, BN, CHID, CIN);

  // token2map
  token2map_scatter<<<BN, blk, 0, stream>>>(bh, loc, bmap);
  map_normalize<<<B_ * HW, blk, 0, stream>>>(bmap);
  smooth_fill<<<B_ * HW, blk, 0, stream>>>(bmap, bfeat);

  // depthwise conv 3x3 + bias
  dwconv3x3<<<B_ * HW, blk, 0, stream>>>(bfeat, dw_w, dw_b, conv);

  // map2token + GELU -> f16
  map2token_gelu<<<BN, blk, 0, stream>>>(conv, loc, t16);

  // fc2: out = t @ fc2_w^T + b   [25088 x 256], K=1024
  wmma_gemm_f16<false><<<dim3(CIN / 64, BN / 128), blk, 0, stream>>>(
      t16, w216, fc2_b, out, BN, CIN, CHID);
}
